// Attention_36232344109507
// MI455X (gfx1250) — matrix operand order repaired and checked by reading, not compiled, not run on hardware
//
#include <hip/hip_runtime.h>
#include <hip/hip_bf16.h>

// ---------------------------------------------------------------------------
// Attention: out = Linear(x); score = out@out^T; attn = softmax(score);
//            context = attn @ x.      B=8, N=2048, H=512, fp32 in/out.
//
// MI455X (gfx1250) strategy:
//  * bf16 WMMA 16x16x32, fp32 accumulate, for all three GEMMs.
//  * Fused attention per 32-query-row block: 32x2048 fp32 score panel in LDS
//    (~264KB of the 320KB/WGP), exact softmax, probabilities converted to
//    bf16 A-fragments on the fly (1/rowsum folded into the cast).
//    M=32 halves L2 panel traffic vs M=16 (each block streams the whole
//    2MB K-panel + 2MB V-panel from L2 once, independent of M).
//  * Register double-buffered B-fragments + cross-k A prefetch so WMMAs
//    don't sit behind s_wait_loadcnt 0 on their own loads.
// ---------------------------------------------------------------------------

typedef __bf16 bf16_t;
typedef bf16_t v16bf __attribute__((ext_vector_type(16)));
typedef bf16_t v8bf  __attribute__((ext_vector_type(8)));
typedef float  v8f   __attribute__((ext_vector_type(8)));

#define B_  8
#define N_  2048
#define H_  512
#define M_  32     // query rows per attention block
#define SLD 2068   // LDS row stride (floats): 2068 % 64 = 20 banks/row skew

static __device__ __forceinline__ v8f wmma_bf16(v16bf a, v16bf b, v8f c) {
  return __builtin_amdgcn_wmma_f32_16x16x32_bf16(
      /*neg_a=*/false, a, /*neg_b=*/false, b,
      /*c_mod=*/(short)0, c, /*reuse_a=*/false, /*reuse_b=*/false);
}

// A-fragment (16x32 bf16, row-major source): lane L -> row (L&15),
// K chunks [k0+8h .. +8) and [k0+16+8h .. +8), h = L>>4.
static __device__ __forceinline__ v16bf load_a_rowmajor(
    const bf16_t* __restrict__ row_ptr, int k0, int h16) {
  union { v16bf v; v8bf h[2]; } a;
  a.h[0] = *(const v8bf*)(row_ptr + k0 + 8 * h16);
  a.h[1] = *(const v8bf*)(row_ptr + k0 + 16 + 8 * h16);
  return a.v;
}

// B-fragment (32x16 bf16): lane L -> column (L&15), K chunks
// [k0+8h .. +8) and [k0+16+8h .. +8), h = L>>4.  `base` is row-major [*, row_stride].
static __device__ __forceinline__ v16bf load_b_colfrag(
    const bf16_t* __restrict__ base, size_t row_stride, int col, int k0,
    int h16) {
  union { v16bf v; v8bf h[2]; } b;
  const bf16_t* p = base + (size_t)col * row_stride + k0;
  b.h[0] = *(const v8bf*)(p + 8 * h16);
  b.h[1] = *(const v8bf*)(p + 16 + 8 * h16);
  return b.v;
}

// ------------------------- casts / transpose -------------------------------
__global__ __launch_bounds__(256) void cast_xw_kernel(
    const float* __restrict__ X, const float* __restrict__ W,
    bf16_t* __restrict__ Xbf, bf16_t* __restrict__ Wbf) {
  size_t i = (size_t)blockIdx.x * blockDim.x + threadIdx.x;  // B*N*H threads
  Xbf[i] = (bf16_t)X[i];
  if (i < (size_t)H_ * H_) Wbf[i] = (bf16_t)W[i];
}

__global__ __launch_bounds__(256) void transpose_x_kernel(
    const float* __restrict__ X, bf16_t* __restrict__ XbfT) {
  // XbfT[b][h][n] = X[b][n][h]; writes coalesced, reads served by L2.
  size_t i = (size_t)blockIdx.x * blockDim.x + threadIdx.x;  // B*H*N threads
  int    n = (int)(i % N_);
  size_t t = i / N_;
  int    h = (int)(t % H_);
  int    b = (int)(t / H_);
  XbfT[i] = (bf16_t)X[((size_t)b * N_ + n) * H_ + h];
}

// ------------------------- O = X @ W^T + bias ------------------------------
// 256 threads = 8 waves; block owns a 16-row strip, each wave 16x64 of O.
__global__ __launch_bounds__(256) void linear_kernel(
    const bf16_t* __restrict__ Xbf,   // [B*N, H]
    const bf16_t* __restrict__ Wbf,   // [H, H]
    const float*  __restrict__ bias,  // [H]
    bf16_t* __restrict__ Obf) {       // [B*N, H]
  const int tid  = threadIdx.x;
  const int wave = tid >> 5;
  const int lane = tid & 31;
  const int h16  = lane >> 4;
  const int l16  = lane & 15;
  const int row0 = blockIdx.x * 16;
  const int col0 = wave * 64;

  v8f acc[4];
#pragma unroll
  for (int t = 0; t < 4; ++t)
#pragma unroll
    for (int j = 0; j < 8; ++j) acc[t][j] = 0.0f;

  const bf16_t* arow = Xbf + (size_t)(row0 + l16) * H_;
  v16bf a = load_a_rowmajor(arow, 0, h16);
  for (int k0 = 0; k0 < H_; k0 += 32) {
    v16bf anext = a;
    if (k0 + 32 < H_) anext = load_a_rowmajor(arow, k0 + 32, h16);
    v16bf bcur = load_b_colfrag(Wbf, H_, col0 + l16, k0, h16);
#pragma unroll
    for (int t = 0; t < 4; ++t) {
      v16bf bnxt = bcur;
      if (t < 3) bnxt = load_b_colfrag(Wbf, H_, col0 + (t + 1) * 16 + l16, k0, h16);
      acc[t] = wmma_bf16(a, bcur, acc[t]);
      bcur = bnxt;
    }
    a = anext;
  }

#pragma unroll
  for (int t = 0; t < 4; ++t) {
    const int col = col0 + t * 16 + l16;
    const float bv = bias[col];
#pragma unroll
    for (int v = 0; v < 8; ++v) {
      const int row = row0 + h16 * 8 + v;
      Obf[(size_t)row * H_ + col] = (bf16_t)(acc[t][v] + bv);
    }
  }
}

// ---------------- fused score -> softmax -> context ------------------------
// One block (16 waves, 512 threads) per (batch, 32-query-row block).
// Wave w: row half rh = w>>3 (16 rows), column strip wc = w&7.
__global__ __launch_bounds__(512) void attn_kernel(
    const bf16_t* __restrict__ Obf,   // [B, N, H]
    const bf16_t* __restrict__ XbfT,  // [B, H, N]
    float* __restrict__ Out) {        // [B, N, H]
  extern __shared__ char smem_raw[];
  float* S       = (float*)smem_raw;   // [M_][SLD] score / prob panel
  float* red     = S + M_ * SLD;       // [M_][16] reduction scratch
  float* rowstat = red + M_ * 16;      // [M_] rowmax then 1/rowsum

  const int tid  = threadIdx.x;
  const int wave = tid >> 5;
  const int lane = tid & 31;
  const int h16  = lane >> 4;
  const int l16  = lane & 15;
  const int rh   = wave >> 3;          // 0/1: which 16-row half
  const int wc   = wave & 7;           // column strip
  const int b    = blockIdx.x >> 6;    // N/32 = 64 row blocks per batch
  const int rb   = blockIdx.x & 63;
  const int row0 = rb * M_;            // first query row of this block

  const bf16_t* Ob = Obf + (size_t)b * N_ * H_;

  // ---- phase 1: S[32][2048] = Oq @ Ok^T ; wave owns 16 rows x 256 keys
  {
    const int colbase = wc * 256;
    v8f acc[16];
#pragma unroll
    for (int t = 0; t < 16; ++t)
#pragma unroll
      for (int j = 0; j < 8; ++j) acc[t][j] = 0.0f;

    const bf16_t* arow = Ob + (size_t)(row0 + rh * 16 + l16) * H_;
    v16bf a = load_a_rowmajor(arow, 0, h16);
    for (int k0 = 0; k0 < H_; k0 += 32) {
      v16bf anext = a;
      if (k0 + 32 < H_) anext = load_a_rowmajor(arow, k0 + 32, h16);
      v16bf bcur = load_b_colfrag(Ob, H_, colbase + l16, k0, h16);
#pragma unroll
      for (int t = 0; t < 16; ++t) {
        v16bf bnxt = bcur;
        if (t < 15)
          bnxt = load_b_colfrag(Ob, H_, colbase + (t + 1) * 16 + l16, k0, h16);
        acc[t] = wmma_bf16(a, bcur, acc[t]);
        bcur = bnxt;
      }
      a = anext;
    }
#pragma unroll
    for (int t = 0; t < 16; ++t) {
      const int col = colbase + t * 16 + l16;
#pragma unroll
      for (int v = 0; v < 8; ++v)
        S[(rh * 16 + h16 * 8 + v) * SLD + col] = acc[t][v];
    }
  }
  __syncthreads();

  // ---- phase 2: exact softmax over 2048 columns (16 threads per row)
  {
    const int r = tid >> 4;   // 0..31
    const int c = tid & 15;
    float m = -3.402823466e38f;
    for (int j = c; j < N_; j += 16) m = fmaxf(m, S[r * SLD + j]);
    red[r * 16 + c] = m;
    __syncthreads();
    if (c == 0) {
      float mm = red[r * 16];
#pragma unroll
      for (int j = 1; j < 16; ++j) mm = fmaxf(mm, red[r * 16 + j]);
      rowstat[r] = mm;
    }
    __syncthreads();
    const float rowmax = rowstat[r];
    float s = 0.0f;
    for (int j = c; j < N_; j += 16) {
      float e = __expf(S[r * SLD + j] - rowmax);
      S[r * SLD + j] = e;
      s += e;
    }
    red[r * 16 + c] = s;
    __syncthreads();
    if (c == 0) {
      float ss = red[r * 16];
#pragma unroll
      for (int j = 1; j < 16; ++j) ss += red[r * 16 + j];
      rowstat[r] = 1.0f / ss;
    }
    __syncthreads();
  }

  // ---- phase 3: ctx[32,512] = P[32,2048] @ X[2048,512]
  // Wave owns rows rh*16..rh*16+15, output columns wc*64..wc*64+63.
  {
    const int col0 = wc * 64;
    v8f acc[4];
#pragma unroll
    for (int t = 0; t < 4; ++t)
#pragma unroll
      for (int j = 0; j < 8; ++j) acc[t][j] = 0.0f;

    const float inv   = rowstat[rh * 16 + l16];  // A rows are m = l16
    const float* srow = S + (size_t)(rh * 16 + l16) * SLD;
    const bf16_t* Xt  = XbfT + (size_t)b * H_ * N_;
    for (int k0 = 0; k0 < N_; k0 += 32) {
      v16bf a;
      const int ka = k0 + 8 * h16;
      const int kb = k0 + 16 + 8 * h16;
#pragma unroll
      for (int j = 0; j < 8; ++j) {
        a[j]     = (bf16_t)(srow[ka + j] * inv);
        a[8 + j] = (bf16_t)(srow[kb + j] * inv);
      }
      v16bf bcur = load_b_colfrag(Xt, N_, col0 + l16, k0, h16);
#pragma unroll
      for (int t = 0; t < 4; ++t) {
        v16bf bnxt = bcur;
        if (t < 3) bnxt = load_b_colfrag(Xt, N_, col0 + (t + 1) * 16 + l16, k0, h16);
        acc[t] = wmma_bf16(a, bcur, acc[t]);
        bcur = bnxt;
      }
    }

#pragma unroll
    for (int t = 0; t < 4; ++t) {
      const int col = col0 + t * 16 + l16;
#pragma unroll
      for (int v = 0; v < 8; ++v) {
        const int row = row0 + rh * 16 + h16 * 8 + v;
        Out[((size_t)b * N_ + row) * H_ + col] = acc[t][v];
      }
    }
  }
}

// ---------------------------------------------------------------------------
extern "C" void kernel_launch(void* const* d_in, const int* in_sizes, int n_in,
                              void* d_out, int out_size, void* d_ws, size_t ws_size,
                              hipStream_t stream) {
  const float* X    = (const float*)d_in[0];  // [B,N,H]
  const float* W    = (const float*)d_in[1];  // [H,H]
  const float* bias = (const float*)d_in[2];  // [H]
  float* out = (float*)d_out;                 // [B,N,H]

  // Workspace carve-up (~48.5 MB total).
  const size_t XB = (size_t)B_ * N_ * H_ * sizeof(bf16_t);  // 16 MiB each
  char* ws = (char*)d_ws;
  bf16_t* Xbf  = (bf16_t*)(ws);
  bf16_t* XbfT = (bf16_t*)(ws + XB);
  bf16_t* Obf  = (bf16_t*)(ws + 2 * XB);
  bf16_t* Wbf  = (bf16_t*)(ws + 3 * XB);

  const int total = B_ * N_ * H_;  // 8,388,608
  cast_xw_kernel<<<total / 256, 256, 0, stream>>>(X, W, Xbf, Wbf);
  transpose_x_kernel<<<total / 256, 256, 0, stream>>>(X, XbfT);
  linear_kernel<<<(B_ * N_) / 16, 256, 0, stream>>>(Xbf, Wbf, bias, Obf);

  // LDS: score panel + reduction scratch (~266 KB of the 320 KB/WGP).
  const size_t smem = (size_t)(M_ * SLD + M_ * 16 + M_) * sizeof(float);
  attn_kernel<<<B_ * (N_ / M_), 512, smem, stream>>>(Obf, XbfT, out);
}
